// MultiHeadSelfAttention_15668040695989
// MI455X (gfx1250) — matrix operand order repaired and checked by reading, not compiled, not run on hardware
//
#include <hip/hip_runtime.h>
#include <hip/hip_bf16.h>

// ---------------- problem constants ----------------
constexpr int kB  = 4;
constexpr int kT  = 2048;
constexpr int kH  = 12;
constexpr int kDK = 64;
constexpr int kDM = 768;   // kH * kDK

typedef __attribute__((ext_vector_type(16))) _Float16 v16h;
typedef __attribute__((ext_vector_type(8)))  _Float16 v8h;
typedef __attribute__((ext_vector_type(8)))  float    v8f;
typedef __attribute__((ext_vector_type(4)))  unsigned int u32x4;
typedef __attribute__((ext_vector_type(8)))  unsigned int u32x8;

// D = A(16x32 f16) * B(32x16 f16) + C(16x16 f32)
__device__ __forceinline__ v8f wmma_f16(v16h a, v16h b, v8f c) {
    return __builtin_amdgcn_wmma_f32_16x16x32_f16(
        /*neg_a=*/false, a, /*neg_b=*/false, b,
        /*c_mod=*/(short)0, c, /*reuse_a=*/false, /*reuse_b=*/false);
}

// A-fragment (16x32, 16-bit): lane row = lane&15; per-lane k values are one
// contiguous 16-element run at kbase, kbase = 32*chunk + 16*(lane>=16).
__device__ __forceinline__ v16h a_frag_f16(const _Float16* rowp, int kbase) {
    union { v16h v; v8h h[2]; } u;
    u.h[0] = *(const v8h*)(rowp + kbase);
    u.h[1] = *(const v8h*)(rowp + kbase + 8);
    return u.v;
}

__device__ __forceinline__ v8h cvt8_f32_to_f16(const float* p) {
    v8h r;
#pragma unroll
    for (int i = 0; i < 8; ++i) r[i] = (_Float16)p[i];
    return r;
}

__device__ __forceinline__ v16h a_frag_f32(const float* rowp, int kbase) {
    union { v16h v; v8h h[2]; } u;
    u.h[0] = cvt8_f32_to_f16(rowp + kbase);
    u.h[1] = cvt8_f32_to_f16(rowp + kbase + 8);
    return u.v;
}

// ---- Tensor Data Mover: 2D tile (rows x 64 f16, row stride 64) -> LDS ----
// D# layout per CDNA5 ISA:
//  group0: [1:0]=count=1, [63:32]=lds_addr, [120:64]=global_addr, [127:126]=type=2
//  group1: [17:16]=data_size=1(2B), [79:48]=tensor_dim0=64, [111:80]=tensor_dim1,
//          [127:112]=tile_dim0=64, [143:128]=tile_dim1=rows, [207:160]=dim0_stride=64
__device__ __forceinline__ void tdm_load_tile(unsigned int lds_byte_addr,
                                              const _Float16* gsrc,
                                              unsigned int rows)
{
    unsigned long long ga = (unsigned long long)(uintptr_t)gsrc;
    u32x4 g0;
    g0[0] = 1u;                                                   // count=1
    g0[1] = lds_byte_addr;                                        // lds_addr
    g0[2] = (unsigned int)ga;                                     // gaddr[31:0]
    g0[3] = (unsigned int)((ga >> 32) & 0x1FFFFFFu) | 0x80000000u; // gaddr[56:32] | type=2
    u32x8 g1;
    g1[0] = 0x00010000u;                 // workgroup_mask=0, data_size=2B
    g1[1] = (unsigned int)kDK << 16;     // tensor_dim0 = 64 (low16)
    g1[2] = (unsigned int)kT  << 16;     // tensor_dim1 = T (low16; high16 in g1[3][15:0]=0)
    g1[3] = (unsigned int)kDK << 16;     // tile_dim0 = 64
    g1[4] = rows;                        // tile_dim1 = rows, tile_dim2 = 0
    g1[5] = (unsigned int)kDK;           // tensor_dim0_stride = 64 (low32)
    g1[6] = 0u;
    g1[7] = 0u;
    asm volatile("tensor_load_to_lds %0, %1"
                 :
                 : "s"(g0), "s"(g1)
                 : "memory");
}

// ---------------- kernel 1: convert weights to f16 ----------------
__global__ __launch_bounds__(256) void cvt_w_kernel(
    const float* __restrict__ Wq, const float* __restrict__ Wk,
    const float* __restrict__ Wv, const float* __restrict__ Wo,
    _Float16* __restrict__ WqH, _Float16* __restrict__ WkH,
    _Float16* __restrict__ WvH, _Float16* __restrict__ WoH)
{
    int i = blockIdx.x * 256 + threadIdx.x;
    if (i < 4096)            WqH[i]          = (_Float16)Wq[i];
    else if (i < 8192)       WkH[i - 4096]   = (_Float16)Wk[i - 4096];
    else if (i < 12288)      WvH[i - 8192]   = (_Float16)Wv[i - 8192];
    else {
        int j = i - 12288;
        if (j < kDM * kDM)   WoH[j]          = (_Float16)Wo[j];
    }
}

// ---------------- kernel 2: per-head QKV projection (WMMA) ----------------
__global__ __launch_bounds__(256) void proj_kernel(
    const float* __restrict__ xq, const float* __restrict__ xk, const float* __restrict__ xv,
    const _Float16* __restrict__ WqH, const _Float16* __restrict__ WkH, const _Float16* __restrict__ WvH,
    const float* __restrict__ bq, const float* __restrict__ bk, const float* __restrict__ bv,
    _Float16* __restrict__ Qh, _Float16* __restrict__ Kh, _Float16* __restrict__ Vh)
{
    const int lane = threadIdx.x & 31;
    const int wave = threadIdx.x >> 5;
    const int w    = blockIdx.x * 8 + wave;     // kB*kH*(kT/16) = 6144 waves
    const int tile = w & 127;                   // kT/16 = 128
    const int h    = (w >> 7) % kH;
    const int b    = w / (128 * kH);
    const int t0   = tile * 16;
    const int hf   = lane >> 4;
    const int lm   = lane & 15;

    const float*    xs[3] = {xq, xk, xv};
    const _Float16* Ws[3] = {WqH, WkH, WvH};
    const float*    bs[3] = {bq, bk, bv};
    _Float16*       Os[3] = {Qh, Kh, Vh};

#pragma unroll
    for (int m = 0; m < 3; ++m) {
        const float* xrow = xs[m] + ((size_t)(b * kT + t0 + lm)) * kDM + h * kDK;
        v16h a0 = a_frag_f32(xrow, 0 * 32 + 16 * hf);
        v16h a1 = a_frag_f32(xrow, 1 * 32 + 16 * hf);
        _Float16* outbase = Os[m] + (((size_t)b * kH + h) * kT + t0) * kDK;
#pragma unroll
        for (int n = 0; n < 4; ++n) {
            const int e = n * 16 + lm;          // output feature (B/D column)
            v16h b0 = *(const v16h*)(Ws[m] + (size_t)e * kDK + 0 * 32 + 16 * hf);
            v16h b1 = *(const v16h*)(Ws[m] + (size_t)e * kDK + 1 * 32 + 16 * hf);
            v8f acc = {};
            acc = wmma_f16(a0, b0, acc);
            acc = wmma_f16(a1, b1, acc);
            const float bias = bs[m][e];
#pragma unroll
            for (int r = 0; r < 8; ++r) {
                const int mr = 8 * hf + r;      // D row
                outbase[(size_t)mr * kDK + e] = (_Float16)(acc[r] + bias);
            }
        }
    }
}

// ---------------- kernel 3: flash attention (WMMA + TDM staging) ----------------
// Block = 8 waves = 128 query rows of one (b,h). Keys processed 32 at a time:
// K tile and V tile are DMA'd into LDS by the Tensor Data Mover (one wave issues,
// waits on TENSORcnt), then V is transposed LDS->LDS for B-fragment loads.
__global__ __launch_bounds__(256) void attn_kernel(
    const _Float16* __restrict__ Qh, const _Float16* __restrict__ Kh,
    const _Float16* __restrict__ Vh, _Float16* __restrict__ Cc)
{
    __shared__ alignas(32) _Float16 sK[32 * 64];    // [key][d]   (TDM dest)
    __shared__ alignas(32) _Float16 sV[32 * 64];    // [key][d]   (TDM dest, staging)
    __shared__ alignas(32) _Float16 sVt[64 * 32];   // [d][key]   (transposed)
    __shared__ alignas(32) _Float16 sP[8][16 * 32]; // per-wave P tile [m][key]

    const int lane = threadIdx.x & 31;
    const int wave = threadIdx.x >> 5;
    const int qt   = blockIdx.x & 15;              // kT/128 = 16
    const int h    = (blockIdx.x >> 4) % kH;
    const int b    = blockIdx.x / (16 * kH);
    const int t0   = qt * 128 + wave * 16;
    const int hf   = lane >> 4;
    const int lm   = lane & 15;

    const size_t bh = (size_t)b * kH + h;
    const _Float16* Qbase = Qh + (bh * kT + t0) * kDK;
    const _Float16* Kbase = Kh + bh * kT * kDK;
    const _Float16* Vbase = Vh + bh * kT * kDK;

    // Q A-fragments, loaded once (row = lm)
    const _Float16* qrow = Qbase + (size_t)lm * kDK;
    const v16h qa0 = a_frag_f16(qrow, 0 * 32 + 16 * hf);
    const v16h qa1 = a_frag_f16(qrow, 1 * 32 + 16 * hf);

    float mrow[8], lrow[8];
    v8f acc[4] = {};
#pragma unroll
    for (int r = 0; r < 8; ++r) { mrow[r] = -1e30f; lrow[r] = 0.0f; }

    const float scale = 0.125f;                     // 1/sqrt(64)

    const unsigned int sK_lds = (unsigned int)(uintptr_t)&sK[0];
    const unsigned int sV_lds = (unsigned int)(uintptr_t)&sV[0];

    // transpose indices (256 threads re-layout 32x64 f16)
    const int ckey = threadIdx.x >> 3;              // 0..31
    const int cdof = (threadIdx.x & 7) * 8;         // 0,8,...,56

    for (int kc = 0; kc < kT / 32; ++kc) {
        __syncthreads();                            // previous tiles fully consumed
        if (wave == 0) {
            tdm_load_tile(sK_lds, Kbase + (size_t)kc * 32 * kDK, 32u);
            tdm_load_tile(sV_lds, Vbase + (size_t)kc * 32 * kDK, 32u);
            __builtin_amdgcn_s_wait_tensorcnt(0);
        }
        __syncthreads();                            // K,V tiles resident in LDS

        // LDS->LDS transpose of V; prefetch next chunk toward L2
        {
            v8h vv = *(const v8h*)(&sV[ckey * 64 + cdof]);
#pragma unroll
            for (int i = 0; i < 8; ++i) sVt[(cdof + i) * 32 + ckey] = vv[i];
            if (kc + 1 < kT / 32) {
                __builtin_prefetch(Kbase + ((size_t)((kc + 1) * 32 + ckey)) * kDK + cdof, 0, 1);
                __builtin_prefetch(Vbase + ((size_t)((kc + 1) * 32 + ckey)) * kDK + cdof, 0, 1);
            }
        }
        __syncthreads();                            // sVt ready

        // S = Q K^T for two 16-key tiles
        v8f s[2];
#pragma unroll
        for (int n = 0; n < 2; ++n) {
            const _Float16* krow = &sK[(n * 16 + lm) * 64]; // B col = key
            v16h kb0 = *(const v16h*)(krow + 0 * 32 + 16 * hf);
            v16h kb1 = *(const v16h*)(krow + 1 * 32 + 16 * hf);
            v8f t = {};
            t = wmma_f16(qa0, kb0, t);
            t = wmma_f16(qa1, kb1, t);
            s[n] = t;
        }

        // online softmax; each (VGPR r, lane-half) is one query row, 16 lanes = 16 keys
#pragma unroll
        for (int r = 0; r < 8; ++r) {
            float a0 = s[0][r] * scale;
            float a1 = s[1][r] * scale;
            float vmax = fmaxf(a0, a1);
            vmax = fmaxf(vmax, __shfl_xor(vmax, 1));
            vmax = fmaxf(vmax, __shfl_xor(vmax, 2));
            vmax = fmaxf(vmax, __shfl_xor(vmax, 4));
            vmax = fmaxf(vmax, __shfl_xor(vmax, 8));
            const float mnew = fmaxf(mrow[r], vmax);
            const float sf   = __expf(mrow[r] - mnew);
            mrow[r] = mnew;
            const float p0 = __expf(a0 - mnew);
            const float p1 = __expf(a1 - mnew);
            s[0][r] = p0;
            s[1][r] = p1;
            float rs = p0 + p1;
            rs += __shfl_xor(rs, 1);
            rs += __shfl_xor(rs, 2);
            rs += __shfl_xor(rs, 4);
            rs += __shfl_xor(rs, 8);
            lrow[r] = lrow[r] * sf + rs;
#pragma unroll
            for (int n = 0; n < 4; ++n) acc[n][r] *= sf;
        }

        // D-layout -> A-layout relayout of P through wave-private LDS
        _Float16* pbase = &sP[wave][0];
#pragma unroll
        for (int r = 0; r < 8; ++r) {
            const int mr = 8 * hf + r;
            pbase[mr * 32 + lm]      = (_Float16)s[0][r];
            pbase[mr * 32 + 16 + lm] = (_Float16)s[1][r];
        }
        const _Float16* prow = pbase + lm * 32;
        union { v16h v; v8h h2[2]; } up;
        up.h2[0] = *(const v8h*)(prow + 16 * hf);
        up.h2[1] = *(const v8h*)(prow + 16 * hf + 8);
        const v16h pa = up.v;

        // O += P * V  (B[k=key][n=d] from transposed sVt: 16 contiguous keys)
#pragma unroll
        for (int n = 0; n < 4; ++n) {
            const _Float16* vrow = &sVt[(n * 16 + lm) * 32];
            v16h vb = *(const v16h*)(vrow + 16 * hf);
            acc[n] = wmma_f16(pa, vb, acc[n]);
        }
    }

    // normalize and write concat [B, T, H*64] (f16)
    _Float16* obase = Cc + ((size_t)b * kT + t0) * kDM + h * kDK;
#pragma unroll
    for (int r = 0; r < 8; ++r) {
        const float inv = 1.0f / lrow[r];
        const int mr = 8 * hf + r;
#pragma unroll
        for (int n = 0; n < 4; ++n)
            obase[(size_t)mr * kDM + n * 16 + lm] = (_Float16)(acc[n][r] * inv);
    }
}

// ---------------- kernel 4: output projection (WMMA, K=768) ----------------
__global__ __launch_bounds__(256) void oproj_kernel(
    const _Float16* __restrict__ Cc, const _Float16* __restrict__ WoH,
    const float* __restrict__ bo, float* __restrict__ out)
{
    const int lane = threadIdx.x & 31;
    const int wave = threadIdx.x >> 5;
    const int w    = blockIdx.x * 8 + wave;    // 512*48 = 24576 waves
    const int n    = w % 48;                   // feature tile
    const int mt   = w / 48;                   // token tile (0..511)
    const int hf   = lane >> 4;
    const int lm   = lane & 15;

    const _Float16* arow = Cc  + ((size_t)mt * 16 + lm) * kDM;
    const int e = n * 16 + lm;
    const _Float16* brow = WoH + (size_t)e * kDM;

    v8f acc = {};
#pragma unroll 4
    for (int c = 0; c < kDM / 32; ++c) {
        v16h a  = a_frag_f16(arow, c * 32 + 16 * hf);
        v16h bf = *(const v16h*)(brow + c * 32 + 16 * hf);
        acc = wmma_f16(a, bf, acc);
    }
    const float bias = bo[e];
    float* ob = out + ((size_t)mt * 16) * kDM;
#pragma unroll
    for (int r = 0; r < 8; ++r)
        ob[(size_t)(8 * hf + r) * kDM + e] = acc[r] + bias;
}

// ---------------- launcher ----------------
extern "C" void kernel_launch(void* const* d_in, const int* in_sizes, int n_in,
                              void* d_out, int out_size, void* d_ws, size_t ws_size,
                              hipStream_t stream)
{
    const float* q  = (const float*)d_in[0];
    const float* k  = (const float*)d_in[1];
    const float* v  = (const float*)d_in[2];
    const float* Wq = (const float*)d_in[3];
    const float* bq = (const float*)d_in[4];
    const float* Wk = (const float*)d_in[5];
    const float* bk = (const float*)d_in[6];
    const float* Wv = (const float*)d_in[7];
    const float* bv = (const float*)d_in[8];
    const float* Wo = (const float*)d_in[9];
    const float* bo = (const float*)d_in[10];
    float* out = (float*)d_out;

    _Float16* ws = (_Float16*)d_ws;
    const size_t nQ = (size_t)kB * kH * kT * kDK;   // == kB*kT*kDM
    _Float16* Qh  = ws;
    _Float16* Kh  = Qh + nQ;
    _Float16* Vh  = Kh + nQ;
    _Float16* Cc  = Vh + nQ;
    _Float16* WqH = Cc + nQ;
    _Float16* WkH = WqH + kDK * kDK;
    _Float16* WvH = WkH + kDK * kDK;
    _Float16* WoH = WvH + kDK * kDK;

    cvt_w_kernel<<<(3 * kDK * kDK + kDM * kDM + 255) / 256, 256, 0, stream>>>(
        Wq, Wk, Wv, Wo, WqH, WkH, WvH, WoH);
    proj_kernel<<<(kB * kH * (kT / 16)) / 8, 256, 0, stream>>>(
        q, k, v, WqH, WkH, WvH, bq, bk, bv, Qh, Kh, Vh);
    attn_kernel<<<kB * kH * (kT / 128), 256, 0, stream>>>(Qh, Kh, Vh, Cc);
    oproj_kernel<<<((kB * kT / 16) * (kDM / 16)) / 8, 256, 0, stream>>>(Cc, WoH, bo, out);
}
